// HandshakingKernel_6614249636254
// MI455X (gfx1250) — matrix operand order repaired and checked by reading, not compiled, not run on hardware
//
#include <hip/hip_runtime.h>
#include <hip/hip_bf16.h>
#include <math.h>

// ---------------------------------------------------------------------------
// Problem constants: B=4, S=512, H=768, V=32
// ---------------------------------------------------------------------------
#define H_DIM   768
#define S_LEN   512
#define B_SZ    4
#define V_WIN   32
#define NCOL    (3 * H_DIM)        // 2304 GEMM output columns (P1|P2|P3)
#define NROW    (B_SZ * S_LEN)     // 2048 GEMM rows
#define NVALID  15888              // #valid (s,j) pairs: 481*32 + 496
#define NV_FULL 15392              // rows with all 32 offsets valid (s<=480)
#define NX      (NROW * H_DIM)     // 1,572,864 elements of X
#define NW      (H_DIM * NCOL)     // 1,769,472 elements of W

#define LDS_ROW 72                 // 64 bf16 payload + 8 pad -> 144B row stride

typedef __attribute__((ext_vector_type(8)))  __bf16 v8bf;
typedef __attribute__((ext_vector_type(16))) __bf16 v16bf;
typedef __attribute__((ext_vector_type(8)))  float  v8f;

#if defined(__has_builtin)
#if __has_builtin(__builtin_amdgcn_global_load_async_to_lds_b128) && \
    __has_builtin(__builtin_amdgcn_s_wait_asynccnt)
#define HAS_ASYNC_LDS 1
#endif
#endif
#ifndef HAS_ASYNC_LDS
#define HAS_ASYNC_LDS 0
#endif

// Builtin wants: (int4 AS1*, int4 AS3*, imm offset, imm cpol)
typedef int v4i_ __attribute__((vector_size(16)));
typedef __attribute__((address_space(1))) v4i_* gv4i_p;
typedef __attribute__((address_space(3))) v4i_* lv4i_p;

// ---------------------------------------------------------------------------
// Kernel 0: one-shot split of X and W into (hi, lo) bf16 pairs so the GEMM
// inner loop carries zero conversion VALU. hi = bf16(x); lo = bf16(x - hi).
// ---------------------------------------------------------------------------
__global__ __launch_bounds__(256) void hs_prep(const float* __restrict__ X,
                                               const float* __restrict__ W,
                                               __bf16* __restrict__ Xh,
                                               __bf16* __restrict__ Xl,
                                               __bf16* __restrict__ Wh,
                                               __bf16* __restrict__ Wl) {
  const int i = blockIdx.x * blockDim.x + threadIdx.x;
  if (i < NX) {
    const float x = X[i];
    const __bf16 h = (__bf16)x;
    Xh[i] = h;
    Xl[i] = (__bf16)(x - (float)h);
  }
  if (i < NW) {
    const float w = W[i];
    const __bf16 h = (__bf16)w;
    Wh[i] = h;
    Wl[i] = (__bf16)(w - (float)h);
  }
}

// ---------------------------------------------------------------------------
// Kernel 1: P[t, n] = sum_h X[t, h] * W[n % 768, (n / 768) * 768 + h]
// bf16x3 split-precision WMMA (A_hi*B_hi + A_hi*B_lo + A_lo*B_hi ~= fp32).
//
// Block = 4 waves = 64(M) x 64(N) tile; all waves share the B tile, which is
// double-buffer staged into LDS in K-chunks of 64 using async global->LDS
// copies (fallback: load + ds_store). Wave w owns M rows [64*bx + 16w, +16).
//
// Fragment layouts per CDNA5 ISA 7.12.2 (wave32):
//   A 16x32 bf16 : lane m=lane&15, half=lane>>4
//                  elems 0..7  -> K = 8*half + e   (contiguous, 16B)
//                  elems 8..15 -> K = 16+8*half+e' (contiguous, 16B)
//   B 32x16 bf16 : lane n=lane&15, elems same K order as A (two 16B pieces)
//   C/D 16x16 f32: lane n=lane&15, vgpr r -> row m = r + 8*half
// LDS B layout: row c (tile column) at byte c*144, payload = 64 bf16 K values.
// Fragment read lanes stride 36 dwords -> 16 distinct banks (conflict-free).
// ---------------------------------------------------------------------------
__global__ __launch_bounds__(128) void hs_gemm(const __bf16* __restrict__ Xh,
                                               const __bf16* __restrict__ Xl,
                                               const __bf16* __restrict__ Wh,
                                               const __bf16* __restrict__ Wl,
                                               float* __restrict__ P) {
  __shared__ __align__(16) __bf16 ldsBh[2][64 * LDS_ROW];
  __shared__ __align__(16) __bf16 ldsBl[2][64 * LDS_ROW];

  const int lane = threadIdx.x & 31;
  const int wave = threadIdx.x >> 5;
  const int half = lane >> 4;
  const int l16  = lane & 15;

  const int m0 = blockIdx.x * 64 + wave * 16;  // this wave's output rows
  const int n0 = blockIdx.y * 64;              // block's output cols (shared)
  const int slice = n0 / H_DIM;                // which of W1/W2/W3 (no straddle)
  const int ocol  = n0 % H_DIM;                // row of W within the slice

  v8f acc[4] = {};  // 4 N-subtiles of 16 cols each

  const size_t aoff = (size_t)(m0 + l16) * H_DIM + 8 * half;
  const __bf16* arh = Xh + aoff;
  const __bf16* arl = Xl + aoff;

  // Per-wave B staging: 4 async b128 per array move 4 rows x 128B each
  // (row = wave*16 + i*4 + lane/8, 16B segment = lane%8).
  const int srow = wave * 16 + (lane >> 3);
  const int sseg = lane & 7;
  const size_t gb0 = (size_t)(ocol + srow) * NCOL + slice * H_DIM + sseg * 8;
  const unsigned lb0 = (unsigned)srow * LDS_ROW + sseg * 8;

  auto stage = [&](int buf, int kc) {
#pragma unroll
    for (int i = 0; i < 4; ++i) {
      const size_t goff = gb0 + (size_t)(i * 4) * NCOL + kc;
      const unsigned loff = lb0 + (unsigned)(i * 4) * LDS_ROW;
#if HAS_ASYNC_LDS
      __builtin_amdgcn_global_load_async_to_lds_b128(
          (gv4i_p)(Wh + goff), (lv4i_p)&ldsBh[buf][loff], 0, 0);
      __builtin_amdgcn_global_load_async_to_lds_b128(
          (gv4i_p)(Wl + goff), (lv4i_p)&ldsBl[buf][loff], 0, 0);
#else
      *(v8bf*)&ldsBh[buf][loff] = *(const v8bf*)(Wh + goff);
      *(v8bf*)&ldsBl[buf][loff] = *(const v8bf*)(Wl + goff);
#endif
    }
  };

  stage(0, 0);

  for (int c = 0; c < H_DIM / 64; ++c) {
    const int kc  = c * 64;
    const int buf = c & 1;
    if (kc + 64 < H_DIM) stage(buf ^ 1, kc + 64);
#if HAS_ASYNC_LDS
    // current chunk's 8 async ops complete once <=8 (next chunk's) remain
    if (kc + 64 < H_DIM) {
      __builtin_amdgcn_s_wait_asynccnt(8);
    } else {
      __builtin_amdgcn_s_wait_asynccnt(0);
    }
#endif
    __syncthreads();  // all waves' staging visible

#pragma unroll
    for (int ks = 0; ks < 2; ++ks) {
      const int k0 = kc + ks * 32;
      // ---- A fragments: two aligned 16B global loads each ----
      const v8bf ah0 = *(const v8bf*)(arh + k0);
      const v8bf ah1 = *(const v8bf*)(arh + k0 + 16);
      const v8bf al0 = *(const v8bf*)(arl + k0);
      const v8bf al1 = *(const v8bf*)(arl + k0 + 16);
      const v16bf a_hi = __builtin_shufflevector(ah0, ah1, 0, 1, 2, 3, 4, 5, 6,
                                                 7, 8, 9, 10, 11, 12, 13, 14, 15);
      const v16bf a_lo = __builtin_shufflevector(al0, al1, 0, 1, 2, 3, 4, 5, 6,
                                                 7, 8, 9, 10, 11, 12, 13, 14, 15);

#pragma unroll
      for (int t = 0; t < 4; ++t) {
        // ---- B fragments from LDS: two 16B ds loads per matrix ----
        const unsigned ro = (unsigned)(16 * t + l16) * LDS_ROW + ks * 32 + half * 8;
        const v8bf bh0 = *(const v8bf*)&ldsBh[buf][ro];
        const v8bf bh1 = *(const v8bf*)&ldsBh[buf][ro + 16];
        const v8bf bl0 = *(const v8bf*)&ldsBl[buf][ro];
        const v8bf bl1 = *(const v8bf*)&ldsBl[buf][ro + 16];
        const v16bf b_hi = __builtin_shufflevector(bh0, bh1, 0, 1, 2, 3, 4, 5, 6,
                                                   7, 8, 9, 10, 11, 12, 13, 14, 15);
        const v16bf b_lo = __builtin_shufflevector(bl0, bl1, 0, 1, 2, 3, 4, 5, 6,
                                                   7, 8, 9, 10, 11, 12, 13, 14, 15);

        v8f cc = acc[t];
        // small correction terms first, dominant term last
        cc = __builtin_amdgcn_wmma_f32_16x16x32_bf16(false, a_hi, false, b_lo,
                                                     (short)0, cc, false, false);
        cc = __builtin_amdgcn_wmma_f32_16x16x32_bf16(false, a_lo, false, b_hi,
                                                     (short)0, cc, false, false);
        cc = __builtin_amdgcn_wmma_f32_16x16x32_bf16(false, a_hi, false, b_hi,
                                                     (short)0, cc, false, false);
        acc[t] = cc;
      }
    }
    __syncthreads();  // done reading buf before it is restaged
  }

  // ---- store D tiles: lanes 0-15 are consecutive columns -> coalesced ----
#pragma unroll
  for (int t = 0; t < 4; ++t) {
#pragma unroll
    for (int r = 0; r < 8; ++r) {
      P[(size_t)(m0 + r + 8 * half) * NCOL + n0 + 16 * t + l16] = acc[t][r];
    }
  }
}

// ---------------------------------------------------------------------------
// Kernel 2: CS3[b, s+1, o] = CS3[b, s, o] + P3[b, s, o]; CS3[b, 0, o] = 0.
// One thread per (b, o); adjacent threads read adjacent addresses.
// ---------------------------------------------------------------------------
__global__ __launch_bounds__(256) void hs_scan(const float* __restrict__ P,
                                               float* __restrict__ C) {
  const int idx = blockIdx.x * blockDim.x + threadIdx.x;
  if (idx >= B_SZ * H_DIM) return;
  const int b = idx / H_DIM;
  const int o = idx % H_DIM;

  const float* p3 = P + (size_t)b * S_LEN * NCOL + 2 * H_DIM + o;
  float* c = C + (size_t)b * (S_LEN + 1) * H_DIM + o;

  float acc = 0.0f;
  c[0] = 0.0f;
  for (int s = 0; s < S_LEN; ++s) {
    acc += p3[(size_t)s * NCOL];
    c[(size_t)(s + 1) * H_DIM] = acc;
  }
}

// ---------------------------------------------------------------------------
// Kernel 3: out[b, r, o] = tanh(P1[b,s] + P2[b,s+j] + (CS3[b,s+j+1]-CS3[b,s])/(j+1) + bias)
// r -> (s, j) inverts np.nonzero's row-major valid-pair ordering:
//   r < 15392       : s = r/32,  j = r%32
//   r >= 15392      : t = s-481, prefix(t) = t*(63-t)/2
// One block per (r, b); 768 outputs per block. Output (195MB > L2) is stored
// non-temporally so it doesn't evict the re-read P/C intermediates from L2.
// ---------------------------------------------------------------------------
__global__ __launch_bounds__(256) void hs_epilogue(const float* __restrict__ P,
                                                   const float* __restrict__ C,
                                                   const float* __restrict__ bias,
                                                   float* __restrict__ out) {
  const int r = blockIdx.x;  // 0..15887
  const int b = blockIdx.y;  // 0..3

  int s, j;
  if (r < NV_FULL) {
    s = r >> 5;
    j = r & 31;
  } else {
    const int rr = r - NV_FULL;
    int t = 0;
    while ((t + 1) * (63 - (t + 1)) / 2 <= rr) ++t;  // t <= 30
    s = 481 + t;
    j = rr - t * (63 - t) / 2;
  }
  const float inv = 1.0f / (float)(j + 1);

  const float* p1 = P + ((size_t)b * S_LEN + s) * NCOL;
  const float* p2 = P + ((size_t)b * S_LEN + s + j) * NCOL + H_DIM;
  const float* ca = C + ((size_t)b * (S_LEN + 1) + s) * H_DIM;
  const float* cb = C + ((size_t)b * (S_LEN + 1) + s + j + 1) * H_DIM;
  float* op = out + ((size_t)b * NVALID + r) * H_DIM;

  for (int o = threadIdx.x; o < H_DIM; o += blockDim.x) {
    const float v = p1[o] + p2[o] + (cb[o] - ca[o]) * inv + bias[o];
    __builtin_nontemporal_store(tanhf(v), &op[o]);
  }
}

// ---------------------------------------------------------------------------
// Launcher
// ---------------------------------------------------------------------------
extern "C" void kernel_launch(void* const* d_in, const int* in_sizes, int n_in,
                              void* d_out, int out_size, void* d_ws, size_t ws_size,
                              hipStream_t stream) {
  const float* X    = (const float*)d_in[0];  // (B, S, H)
  const float* W    = (const float*)d_in[1];  // (H, 3H)
  const float* bias = (const float*)d_in[2];  // (H,)
  float* out = (float*)d_out;                 // (B, NVALID, H)

  // Workspace layout (floats then bf16 arrays), ~38.5 MB total:
  float* P = (float*)d_ws;                        // (NROW, NCOL)   18.9 MB
  float* C = P + (size_t)NROW * NCOL;             // (B, S+1, H)     6.3 MB
  __bf16* Xh = (__bf16*)(C + (size_t)B_SZ * (S_LEN + 1) * H_DIM);
  __bf16* Xl = Xh + (size_t)NX;
  __bf16* Wh = Xl + (size_t)NX;
  __bf16* Wl = Wh + (size_t)NW;

  {
    const int n = (NW > NX ? NW : NX);            // 1,769,472
    hs_prep<<<(n + 255) / 256, 256, 0, stream>>>(X, W, Xh, Xl, Wh, Wl);
  }
  {
    dim3 grid(NROW / 64, NCOL / 64);              // (32, 36)
    hs_gemm<<<grid, 128, 0, stream>>>(Xh, Xl, Wh, Wl, P);
  }
  {
    const int n = B_SZ * H_DIM;                   // 3072
    hs_scan<<<(n + 255) / 256, 256, 0, stream>>>(P, C);
  }
  {
    dim3 grid(NVALID, B_SZ);
    hs_epilogue<<<grid, 256, 0, stream>>>(P, C, bias, out);
  }
}
